// ConvLSTM_62929860821477
// MI455X (gfx1250) — matrix operand order repaired and checked by reading, not compiled, not run on hardware
//
#include <hip/hip_runtime.h>
#include <hip/hip_bf16.h>

// ---------------------------------------------------------------------------
// ConvLSTM (2-layer, T=16) on gfx1250 via implicit shift-GEMM + WMMA f16.
//   B=4, T=16, C=1, H=W=64, HID=64, conv 3x3 SAME, gates order i,f,o,g.
// Layer0: K = 96 padded channels  (0..63 = h0, 64 = x, 65..95 = zero)
// Layer1: K = 128 channels        (0..63 = h0, 64..127 = h1)
// ---------------------------------------------------------------------------

typedef __attribute__((ext_vector_type(16))) _Float16 v16h;
typedef __attribute__((ext_vector_type(8)))  _Float16 v8h;
typedef __attribute__((ext_vector_type(8)))  float    v8f;
typedef __attribute__((ext_vector_type(4)))  int      v4i;

#define BATCH 4
#define TSTEPS 16
#define HID 64
#define HDIM 64
#define WDIM 64
#define MGATES 256              // 4*HID output channels per conv

#if __has_builtin(__builtin_amdgcn_global_load_async_to_lds_b128) && \
    __has_builtin(__builtin_amdgcn_s_wait_asynccnt)
#define USE_ASYNC_LDS 1
#else
#define USE_ASYNC_LDS 0
#endif

// 16-byte global -> LDS copy; async (ASYNCcnt-tracked) when available.
// Builtin signature (from hipcc diagnostic): param0 = AS1 int4*, so pass
// exactly-typed global/LDS int4 pointers.
__device__ __forceinline__ void copy16_to_lds(const _Float16* g, _Float16* l) {
#if USE_ASYNC_LDS
    __builtin_amdgcn_global_load_async_to_lds_b128(
        (__attribute__((address_space(1))) v4i*)g,
        (__attribute__((address_space(3))) v4i*)l, 0, 0);
#else
    *(v8h*)l = *(const v8h*)g;
#endif
}

__device__ __forceinline__ void fill_wait_async() {
#if USE_ASYNC_LDS
    __builtin_amdgcn_s_wait_asynccnt(0);
#endif
}

// fast tanh on the TRANS pipe: tanh(v) = 1 - 2/(exp(2v)+1)
__device__ __forceinline__ float fast_tanh(float v) {
    float e = __expf(2.f * v);
    return 1.f - 2.f / (e + 1.f);
}
__device__ __forceinline__ float fast_sigmoid(float v) {
    return 1.f / (1.f + __expf(-v));
}

// ---------------------------------------------------------------------------
// Weight packing: rearrange [256][Cin][3][3] f32 into per-lane WMMA A-fragment
// order (f16).  Layout: aw[(kk*KC + kc)][mtile][lane][j], 16 f16 per lane.
// A-matrix 16x32 f16 per-lane map (wave32):
//   m      = mtile*16 + (lane & 15)
//   khalf  = lane >> 4
//   K(j)   = khalf*16 + j
// ---------------------------------------------------------------------------
__global__ void pack_weights_kernel(const float* __restrict__ w,
                                    _Float16* __restrict__ aw,
                                    int KC, int cin, int is_l0, int total) {
    int idx = blockIdx.x * blockDim.x + threadIdx.x;
    if (idx >= total) return;
    int j     = idx & 15;
    int lane  = (idx >> 4) & 31;
    int mtile = (idx >> 9) & 15;
    int rest  = idx >> 13;
    int kc    = rest % KC;
    int kk    = rest / KC;
    int ky = kk / 3, kx = kk % 3;
    int m     = mtile * 16 + (lane & 15);
    int khalf = lane >> 4;
    int K     = khalf * 16 + j;
    int cp    = kc * 32 + K;           // packed channel
    int orig;
    if (is_l0) {                       // reference concat([x, h]): ch0 = x
        orig = (cp < 64) ? (cp + 1) : ((cp == 64) ? 0 : -1);
    } else {
        orig = (cp < cin) ? cp : -1;
    }
    float val = 0.f;
    if (orig >= 0)
        val = w[((m * cin + orig) * 3 + ky) * 3 + kx];
    aw[idx] = (_Float16)val;
}

// ---------------------------------------------------------------------------
// One ConvLSTM cell step for one layer.
// Grid: 256 WGs = (b,y).  Block: 256 threads = 8 wave32.
// Wave w: hid-tile ht = w>>1 (M-tiles ht+4g, g=gate), N-half nh = w&1.
// ---------------------------------------------------------------------------
template <int CINP, int KC, bool IS_L1>
__global__ __launch_bounds__(256)
void cell_kernel(const float* __restrict__ x,      // layer0: x [B][T][1][H][W]
                 int t,
                 const _Float16* __restrict__ hA,  // channel-last [B][H][W][64]
                 const _Float16* __restrict__ hB,  // layer1 only
                 const _Float16* __restrict__ aw,  // packed weights
                 const float* __restrict__ bias,   // [256]
                 float* __restrict__ cstate,       // [B][H][W][64] f32, in/out
                 _Float16* __restrict__ hout,      // [B][H][W][64] f16
                 float* __restrict__ out) {        // layer1: [B][T][64][H][W]
    __shared__ _Float16 lds_in[3 * 66 * CINP];
    __shared__ float    lds_bias[MGATES];

    const int tid  = threadIdx.x;
    const int b    = blockIdx.x >> 6;
    const int y    = blockIdx.x & 63;
    const int wave = tid >> 5;
    const int lane = tid & 31;

    lds_bias[tid] = bias[tid];

    // ---- stage halo tile [3 rows][66 cols][CINP ch] into LDS ---------------
    // One thread per (row, col) pair: 198 pairs; copies channel-contiguous
    // 16B chunks straight into LDS (async-to-LDS when the builtin exists).
    if (tid < 3 * 66) {
        const int r   = tid / 66;            // 0..2
        const int col = tid - r * 66;        // 0..65
        const int xx  = col - 1;
        const int yy  = y + r - 1;
        _Float16* lrow = lds_in + (r * 66 + col) * CINP;
        if (xx >= 0 && xx < WDIM && yy >= 0 && yy < HDIM) {
            const int p = (b * HDIM + yy) * WDIM + xx;
            const _Float16* ga = hA + p * 64;
#pragma unroll
            for (int c8 = 0; c8 < 8; ++c8)
                copy16_to_lds(ga + c8 * 8, lrow + c8 * 8);
            if constexpr (IS_L1) {
                const _Float16* gb = hB + p * 64;
#pragma unroll
                for (int c8 = 0; c8 < 8; ++c8)
                    copy16_to_lds(gb + c8 * 8, lrow + 64 + c8 * 8);
            } else {
                v8h vx = {};
                vx[0] = (_Float16)x[((b * TSTEPS + t) << 12) + (yy << 6) + xx];
                *(v8h*)(lrow + 64) = vx;
                v8h z = {};
                *(v8h*)(lrow + 72) = z;
                *(v8h*)(lrow + 80) = z;
                *(v8h*)(lrow + 88) = z;
            }
        } else {
            v8h z = {};
#pragma unroll
            for (int c8 = 0; c8 < CINP / 8; ++c8)
                *(v8h*)(lrow + c8 * 8) = z;
        }
    }
    fill_wait_async();
    __syncthreads();

    // ---- implicit shift-GEMM: 9 taps x KC chunks of K=32 -------------------
    const int ht = wave >> 1;   // hid 16-tile: 0..3
    const int nh = wave & 1;    // N half: 0..1

    v8f acc[4][2] = {};         // [gate][ntile], 16x16 f32 tiles

    for (int kk = 0; kk < 9; ++kk) {
        const int kyr = kk / 3;     // LDS row
        const int kxs = kk % 3;     // column shift
        for (int kc = 0; kc < KC; ++kc) {
            const _Float16* awb = aw + (size_t)(kk * KC + kc) * (16 * 512);
            if (kk * KC + kc + 1 < 9 * KC)
                __builtin_prefetch(awb + 16 * 512, 0, 0);   // global_prefetch

            v16h afr[4];
#pragma unroll
            for (int g = 0; g < 4; ++g) {
                int mtile = ht + g * 4;
                afr[g] = *(const v16h*)(awb + mtile * 512 + lane * 16);
            }
            v16h bfr[2];
#pragma unroll
            for (int ni = 0; ni < 2; ++ni) {
                int colx = (nh * 2 + ni) * 16 + (lane & 15) + kxs;
                bfr[ni] = *(const v16h*)(lds_in + (kyr * 66 + colx) * CINP +
                                         kc * 32 + ((lane >> 4) << 4));
            }
#pragma unroll
            for (int g = 0; g < 4; ++g)
#pragma unroll
                for (int ni = 0; ni < 2; ++ni)
                    acc[g][ni] = __builtin_amdgcn_wmma_f32_16x16x32_f16(
                        false, afr[g], false, bfr[ni],
                        (short)0, acc[g][ni], false, false);
        }
    }

    // ---- fused LSTM gate math (all gates of a hid channel live in-lane) ----
    const int hb = ht * 16 + ((lane >> 4) << 3);   // hid channel base (+v)
#pragma unroll
    for (int ni = 0; ni < 2; ++ni) {
        const int xpix = (nh * 2 + ni) * 16 + (lane & 15);
        const int p    = (b * HDIM + y) * WDIM + xpix;
        v8f cp = *(const v8f*)(cstate + p * 64 + hb);
        v8f cn;
        v8h hv;
        float hf[8];
#pragma unroll
        for (int v = 0; v < 8; ++v) {
            const int hc = hb + v;
            float iv = acc[0][ni][v] + lds_bias[hc];
            float fv = acc[1][ni][v] + lds_bias[64 + hc];
            float ov = acc[2][ni][v] + lds_bias[128 + hc];
            float gv = acc[3][ni][v] + lds_bias[192 + hc];
            float si = fast_sigmoid(iv);
            float sf = fast_sigmoid(fv);
            float so = fast_sigmoid(ov);
            float cv = sf * cp[v] + si * fast_tanh(gv);
            float hh = so * fast_tanh(cv);
            cn[v] = cv;
            hv[v] = (_Float16)hh;
            hf[v] = hh;
        }
        *(v8f*)(cstate + p * 64 + hb) = cn;
        *(v8h*)(hout + p * 64 + hb) = hv;
        if constexpr (IS_L1) {
#pragma unroll
            for (int v = 0; v < 8; ++v)
                out[(size_t)(((b * TSTEPS + t) * 64 + hb + v) << 12) +
                    (y << 6) + xpix] = hf[v];
        }
    }
}

// ---------------------------------------------------------------------------
// Host launcher (graph-capture safe: only async ops on `stream`).
// ---------------------------------------------------------------------------
extern "C" void kernel_launch(void* const* d_in, const int* in_sizes, int n_in,
                              void* d_out, int out_size, void* d_ws, size_t ws_size,
                              hipStream_t stream) {
    const float* x  = (const float*)d_in[0];
    const float* w0 = (const float*)d_in[1];
    const float* b0 = (const float*)d_in[2];
    const float* w1 = (const float*)d_in[3];
    const float* b1 = (const float*)d_in[4];
    float* out = (float*)d_out;

    const size_t HSZ = (size_t)BATCH * HDIM * WDIM * HID;   // state elements
    size_t off = 0;
    auto carve = [&](size_t bytes) {
        char* p = (char*)d_ws + off;
        off += (bytes + 255) & ~(size_t)255;
        return (void*)p;
    };
    _Float16* h0a = (_Float16*)carve(HSZ * sizeof(_Float16));
    _Float16* h0b = (_Float16*)carve(HSZ * sizeof(_Float16));
    _Float16* h1a = (_Float16*)carve(HSZ * sizeof(_Float16));
    _Float16* h1b = (_Float16*)carve(HSZ * sizeof(_Float16));
    float*    c0  = (float*)carve(HSZ * sizeof(float));
    float*    c1  = (float*)carve(HSZ * sizeof(float));
    const int TOT0 = 9 * 3 * (16 * 512);    // layer0 packed f16 count
    const int TOT1 = 9 * 4 * (16 * 512);    // layer1 packed f16 count
    _Float16* aw0 = (_Float16*)carve((size_t)TOT0 * sizeof(_Float16));
    _Float16* aw1 = (_Float16*)carve((size_t)TOT1 * sizeof(_Float16));
    (void)ws_size; (void)in_sizes; (void)n_in; (void)out_size;

    // pack weights into WMMA A-fragment order (f32 -> f16)
    pack_weights_kernel<<<(TOT0 + 255) / 256, 256, 0, stream>>>(w0, aw0, 3, 65, 1, TOT0);
    pack_weights_kernel<<<(TOT1 + 255) / 256, 256, 0, stream>>>(w1, aw1, 4, 128, 0, TOT1);

    // zero recurrent state (re-done every call -> deterministic under replay)
    (void)hipMemsetAsync(h0a, 0, HSZ * sizeof(_Float16), stream);
    (void)hipMemsetAsync(h0b, 0, HSZ * sizeof(_Float16), stream);
    (void)hipMemsetAsync(h1a, 0, HSZ * sizeof(_Float16), stream);
    (void)hipMemsetAsync(h1b, 0, HSZ * sizeof(_Float16), stream);
    (void)hipMemsetAsync(c0, 0, HSZ * sizeof(float), stream);
    (void)hipMemsetAsync(c1, 0, HSZ * sizeof(float), stream);

    dim3 grid(BATCH * HDIM), block(256);
    for (int t = 0; t < TSTEPS; ++t) {
        const _Float16* h0r = (t & 1) ? h0b : h0a;
        _Float16*       h0w = (t & 1) ? h0a : h0b;
        const _Float16* h1r = (t & 1) ? h1b : h1a;
        _Float16*       h1w = (t & 1) ? h1a : h1b;
        cell_kernel<96, 3, false><<<grid, block, 0, stream>>>(
            x, t, h0r, nullptr, aw0, b0, c0, h0w, nullptr);
        cell_kernel<128, 4, true><<<grid, block, 0, stream>>>(
            nullptr, t, h0w, h1r, aw1, b1, c1, h1w, out);
    }
}
